// QMixingNetwork_67310727463653
// MI455X (gfx1250) — matrix operand order repaired and checked by reading, not compiled, not run on hardware
//
#include <hip/hip_runtime.h>
#include <hip/hip_bf16.h>
#include <math.h>

// ---------------------------------------------------------------------------
// QMIX mixing network for MI455X (gfx1250, wave32, WMMA 16x16x32 bf16).
//   B=32768, STATE_DIM=512, EMBED=64, N_AGENTS=32
// Compute-bound vs HBM (~250 FLOP/byte); the real limiter is L2 traffic for
// the per-sample hypernet weights, so each block processes a 64-row batch
// tile (4 x 16-row m-tiles) and reuses every B operand 4x from registers.
// ---------------------------------------------------------------------------

#define SDIM    512
#define EMB     64
#define NAG     32
#define MB      64      // batch rows per block
#define MT      4       // 16-row m-tiles per block

// Padded LDS strides (elements) for bank-conflict-free ds_read_b128 / b32:
#define A_STR   520     // bf16; 1040B row stride -> 4-bank skew per row
#define H_STR   72      // bf16; 144B  row stride -> 36-dword skew per row
#define F_STR   65      // f32;  1-bank skew per row
#define Q_STR   33      // f32 qs stride

typedef __bf16 bf16_t;
typedef bf16_t v16bf __attribute__((ext_vector_type(16)));
typedef bf16_t v8bf  __attribute__((ext_vector_type(8)));
typedef float  v8f   __attribute__((ext_vector_type(8)));

__device__ __forceinline__ v8f wmma_bf16(v16bf a, v16bf b, v8f c) {
  return __builtin_amdgcn_wmma_f32_16x16x32_bf16(
      false, a, false, b, (short)0, c, false, false);
}

__device__ __forceinline__ v8f v8f_zero() {
  return (v8f){0.f,0.f,0.f,0.f,0.f,0.f,0.f,0.f};
}

// Load a 16x32 bf16 A operand from LDS (row-major, stride H_STR or A_STR).
// Lane layout (ISA 7.12.2): lanes 0-15 row M=lane, lanes 16-31 same rows,
// K-halves selected by lane>>4; two contiguous 8-elem chunks per lane.
__device__ __forceinline__ v16bf load_a_lds(const bf16_t* base, int stride,
                                            int row, int kbase, int ksel) {
  const bf16_t* p = base + row * stride + kbase + ksel * 8;
  v8bf c0 = *(const v8bf*)(p);
  v8bf c1 = *(const v8bf*)(p + 16);
  v16bf a;
#pragma unroll
  for (int j = 0; j < 8; ++j) { a[j] = c0[j]; a[8 + j] = c1[j]; }
  return a;
}

// ---------------------------------------------------------------------------
// Pack fp32 W[O,Kin] (used as x @ W.T, so B = W^T) into the WMMA bf16
// B-operand lane layout; each lane's 16 values contiguous (one 32B load).
// ---------------------------------------------------------------------------
__global__ void pack_b_kernel(const float* __restrict__ W, bf16_t* __restrict__ out,
                              int n_tiles, int ksteps, int Kin) {
  int idx   = blockIdx.x * blockDim.x + threadIdx.x;   // tile_lin*32 + lane
  int total = n_tiles * ksteps * 32;
  if (idx >= total) return;
  int lane     = idx & 31;
  int tile_lin = idx >> 5;
  int kk       = tile_lin % ksteps;
  int n_tile   = tile_lin / ksteps;
  int n        = n_tile * 16 + (lane & 15);
  int kbase    = kk * 32 + (lane >> 4) * 8;
  bf16_t* dst  = out + (size_t)idx * 16;
#pragma unroll
  for (int j = 0; j < 16; ++j)
    dst[j] = (bf16_t)W[(size_t)n * Kin + kbase + j + (j >= 8 ? 8 : 0)];
}

// ---------------------------------------------------------------------------
// Main fused kernel: one block = 128 threads = 4 waves = 64 batch rows.
// ---------------------------------------------------------------------------
__global__ __launch_bounds__(128) void qmix_kernel(
    const float* __restrict__ agent_qs,   // [B, 32]
    const float* __restrict__ states,     // [B, 512]
    const bf16_t* __restrict__ packProj,  // 4 x [4 ntile][16 kstep][32 lane][16]
    const bf16_t* __restrict__ packW1L2,  // [128][2][32][16]
    const bf16_t* __restrict__ packW2L2,  // [4][2][32][16]
    const float* __restrict__ w1_l1_b, const float* __restrict__ w1_l2_b,
    const float* __restrict__ w2_l1_b, const float* __restrict__ w2_l2_b,
    const float* __restrict__ b1_b,    const float* __restrict__ b2_l1_b,
    const float* __restrict__ b2_l2_w, const float* __restrict__ b2_l2_b,
    float* __restrict__ out)              // [B]
{
  __shared__ bf16_t lds_A[MB * A_STR];        // ~65 KB  states as bf16
  __shared__ float  lds_qs[MB * Q_STR];       // ~8.3 KB
  __shared__ bf16_t lds_h1[MB * H_STR];       // 9 KB
  __shared__ bf16_t lds_h2[MB * H_STR];       // 9 KB
  __shared__ float  lds_b1[MB * F_STR];       // ~16.3 KB
  __shared__ float  lds_hb[MB * F_STR];       // ~16.3 KB
  __shared__ float  lds_w2[MB * F_STR];       // ~16.3 KB
  __shared__ float  lds_hidden[MB * F_STR];   // ~16.3 KB   (~157 KB total)

  const int tid  = threadIdx.x;
  const int wave = tid >> 5;        // 0..3
  const int lane = tid & 31;
  const int mrow = lane & 15;
  const int ksel = lane >> 4;
  const int row0 = blockIdx.x * MB;

  // ---- Stage 0: stage states (f32->bf16) + qs into LDS; zero hidden -------
  for (int i = tid * 8; i < MB * SDIM; i += 128 * 8) {
    int r = i >> 9;          // /512
    int c = i & (SDIM - 1);
    v8f f = *(const v8f*)(states + (size_t)(row0 + r) * SDIM + c);
    v8bf b;
#pragma unroll
    for (int j = 0; j < 8; ++j) b[j] = (bf16_t)f[j];
    *(v8bf*)(lds_A + r * A_STR + c) = b;
  }
  for (int i = tid; i < MB * NAG; i += 128) {
    int r = i >> 5, a = i & 31;
    lds_qs[r * Q_STR + a] = agent_qs[(size_t)(row0 + r) * NAG + a];
  }
  for (int i = tid; i < MB * F_STR; i += 128) lds_hidden[i] = 0.f;
  __syncthreads();

  const int colw = wave * 16 + mrow;   // N column owned by this lane (per tile)

  // ---- Stage 1: four state->64 projections, B reused across 4 m-tiles -----
  v8f acc[MT][4];
#pragma unroll
  for (int mt = 0; mt < MT; ++mt)
#pragma unroll
    for (int m = 0; m < 4; ++m) acc[mt][m] = v8f_zero();

  for (int kk = 0; kk < 16; ++kk) {
    v16bf bmat[4];
#pragma unroll
    for (int m = 0; m < 4; ++m)
      bmat[m] = *(const v16bf*)(packProj +
          ((((size_t)m * 4 + wave) * 16 + kk) * 32 + lane) * 16);
#pragma unroll
    for (int mt = 0; mt < MT; ++mt) {
      v16bf a = load_a_lds(lds_A, A_STR, mt * 16 + mrow, kk * 32, ksel);
#pragma unroll
      for (int m = 0; m < 4; ++m)
        acc[mt][m] = wmma_bf16(a, bmat[m], acc[mt][m]);
    }
  }
  {
    const float bh1 = w1_l1_b[colw];
    const float bb1 = b1_b[colw];
    const float bh2 = w2_l1_b[colw];
    const float bhb = b2_l1_b[colw];
#pragma unroll
    for (int mt = 0; mt < MT; ++mt)
#pragma unroll
      for (int r = 0; r < 8; ++r) {
        const int row = mt * 16 + r + 8 * ksel;
        lds_h1[row * H_STR + colw] = (bf16_t)fmaxf(acc[mt][0][r] + bh1, 0.f);
        lds_b1[row * F_STR + colw] = acc[mt][1][r] + bb1;
        lds_h2[row * H_STR + colw] = (bf16_t)fmaxf(acc[mt][2][r] + bh2, 0.f);
        lds_hb[row * F_STR + colw] = fmaxf(acc[mt][3][r] + bhb, 0.f);
      }
  }
  __syncthreads();

  // ---- Stage 2: w1 = |h1 @ W1L2^T + b| fused with agent mix ----------------
  // wave w owns agents [8w,8w+8); per agent hold B for all 4 e-subtiles in
  // regs (64 VGPRs) and replay against the 4 m-tiles.
  {
    v8f hidacc[MT][4];
#pragma unroll
    for (int mt = 0; mt < MT; ++mt)
#pragma unroll
      for (int e = 0; e < 4; ++e) hidacc[mt][e] = v8f_zero();

    for (int aa = 0; aa < 8; ++aa) {
      const int agent = wave * 8 + aa;
      v16bf ball[4][2];
      float bias[4];
#pragma unroll
      for (int e16 = 0; e16 < 4; ++e16) {
        const int n_tile = agent * 4 + e16;      // w1 flat col tile: a*64+e
#pragma unroll
        for (int s = 0; s < 2; ++s)
          ball[e16][s] = *(const v16bf*)(packW1L2 +
              (((size_t)n_tile * 2 + s) * 32 + lane) * 16);
        bias[e16] = w1_l2_b[n_tile * 16 + mrow];
      }
#pragma unroll
      for (int mt = 0; mt < MT; ++mt) {
        v16bf a0 = load_a_lds(lds_h1, H_STR, mt * 16 + mrow, 0, ksel);
        v16bf a1 = load_a_lds(lds_h1, H_STR, mt * 16 + mrow, 32, ksel);
        float q[8];
#pragma unroll
        for (int r = 0; r < 8; ++r)
          q[r] = lds_qs[(mt * 16 + r + 8 * ksel) * Q_STR + agent];
#pragma unroll
        for (int e16 = 0; e16 < 4; ++e16) {
          v8f c = v8f_zero();
          c = wmma_bf16(a0, ball[e16][0], c);
          c = wmma_bf16(a1, ball[e16][1], c);
#pragma unroll
          for (int r = 0; r < 8; ++r)
            hidacc[mt][e16][r] += fabsf(c[r] + bias[e16]) * q[r];
        }
      }
    }
#pragma unroll
    for (int mt = 0; mt < MT; ++mt)
#pragma unroll
      for (int e16 = 0; e16 < 4; ++e16)
#pragma unroll
        for (int r = 0; r < 8; ++r)
          atomicAdd(&lds_hidden[(mt * 16 + r + 8 * ksel) * F_STR + e16 * 16 + mrow],
                    hidacc[mt][e16][r]);
  }

  // ---- Stage 3: w2 = |h2 @ W2L2^T + b|, wave w owns N-tile w ---------------
  {
    v16bf b0 = *(const v16bf*)(packW2L2 + (((size_t)wave * 2 + 0) * 32 + lane) * 16);
    v16bf b1v = *(const v16bf*)(packW2L2 + (((size_t)wave * 2 + 1) * 32 + lane) * 16);
    const float bias = w2_l2_b[colw];
#pragma unroll
    for (int mt = 0; mt < MT; ++mt) {
      v16bf a0 = load_a_lds(lds_h2, H_STR, mt * 16 + mrow, 0, ksel);
      v16bf a1 = load_a_lds(lds_h2, H_STR, mt * 16 + mrow, 32, ksel);
      v8f c = v8f_zero();
      c = wmma_bf16(a0, b0, c);
      c = wmma_bf16(a1, b1v, c);
#pragma unroll
      for (int r = 0; r < 8; ++r)
        lds_w2[(mt * 16 + r + 8 * ksel) * F_STR + colw] = fabsf(c[r] + bias);
    }
  }
  __syncthreads();

  // ---- Stage 4: epilogue: elu, dot with w2, b2 tail ------------------------
  if (tid < MB) {
    const int row = tid;
    float qv = 0.f;
    float b2 = b2_l2_b[0];
#pragma unroll 8
    for (int e = 0; e < EMB; ++e) {
      float h  = lds_hidden[row * F_STR + e] + lds_b1[row * F_STR + e];
      float eh = (h > 0.f) ? h : (__expf(h) - 1.f);
      qv += eh * lds_w2[row * F_STR + e];
      b2 += lds_hb[row * F_STR + e] * b2_l2_w[e];
    }
    out[row0 + row] = qv + b2;
  }
}

// ---------------------------------------------------------------------------
// Launch: pack weights into WMMA B layout (tiny), then the fused kernel.
//   packProj : 4 * 32768 bf16 = 256 KB
//   packW1L2 : 131072 bf16    = 256 KB
//   packW2L2 : 4096 bf16      = 8 KB
// ---------------------------------------------------------------------------
extern "C" void kernel_launch(void* const* d_in, const int* in_sizes, int n_in,
                              void* d_out, int out_size, void* d_ws, size_t ws_size,
                              hipStream_t stream) {
  const float* agent_qs = (const float*)d_in[0];
  const float* states   = (const float*)d_in[1];
  const float* w1_l1_w  = (const float*)d_in[2];
  const float* w1_l1_b  = (const float*)d_in[3];
  const float* w1_l2_w  = (const float*)d_in[4];
  const float* w1_l2_b  = (const float*)d_in[5];
  const float* w2_l1_w  = (const float*)d_in[6];
  const float* w2_l1_b  = (const float*)d_in[7];
  const float* w2_l2_w  = (const float*)d_in[8];
  const float* w2_l2_b  = (const float*)d_in[9];
  const float* b1_w     = (const float*)d_in[10];
  const float* b1_b     = (const float*)d_in[11];
  const float* b2_l1_w  = (const float*)d_in[12];
  const float* b2_l1_b  = (const float*)d_in[13];
  const float* b2_l2_w  = (const float*)d_in[14];
  const float* b2_l2_b  = (const float*)d_in[15];
  float* out = (float*)d_out;

  bf16_t* packProj = (bf16_t*)d_ws;
  bf16_t* packW1L2 = packProj + 131072;
  bf16_t* packW2L2 = packW1L2 + 131072;

  const int B = in_sizes[0] / NAG;   // 32768

  const float* projW[4] = { w1_l1_w, b1_w, w2_l1_w, b2_l1_w };
  for (int m = 0; m < 4; ++m)
    pack_b_kernel<<<16, 128, 0, stream>>>(projW[m], packProj + (size_t)m * 32768,
                                          4, 16, SDIM);
  pack_b_kernel<<<64, 128, 0, stream>>>(w1_l2_w, packW1L2, 128, 2, EMB);
  pack_b_kernel<<<2, 128, 0, stream>>>(w2_l2_w, packW2L2, 4, 2, EMB);

  const int nblocks = B / MB;        // 512
  qmix_kernel<<<nblocks, 128, 0, stream>>>(
      agent_qs, states, packProj, packW1L2, packW2L2,
      w1_l1_b, w1_l2_b, w2_l1_b, w2_l2_b, b1_b, b2_l1_b, b2_l2_w, b2_l2_b, out);
}
